// GNNEncoder_18511309046197
// MI455X (gfx1250) — matrix operand order repaired and checked by reading, not compiled, not run on hardware
//
#include <hip/hip_runtime.h>

#define Hdim 128
#define LDS_STRIDE 136   // pad 128 bf16 row to reduce LDS bank conflicts

typedef __attribute__((ext_vector_type(16))) __bf16 v16bf;
typedef __attribute__((ext_vector_type(8)))  float  v8f;

union BFrag { v16bf v; unsigned int u[8]; };

__device__ __forceinline__ unsigned int f2bf(float f) {
  // round-to-nearest-even fp32 -> bf16 (returned in low 16 bits)
  unsigned int u = __float_as_uint(f);
  return (u + 0x7FFFu + ((u >> 16) & 1u)) >> 16;
}

// ---------------------------------------------------------------------------
// Generic  C[N x 128] = A[N x 128] @ W[128 x 128]^T + bias (+ resid)
// bf16 WMMA, f32 accumulate. One block = 128-row tile, 8 waves, wave = 16 rows.
// B fragments double-buffered so ds_load overlaps v_wmma.
// ---------------------------------------------------------------------------
__global__ __launch_bounds__(256)
void gemm_wmma(const float* __restrict__ A, const float* __restrict__ W,
               const float* __restrict__ bias, const float* __restrict__ resid,
               float* __restrict__ C, int N)
{
  __shared__ unsigned short Wl[Hdim * LDS_STRIDE];  // Wl[n][k] = bf16(W[n][k])
  __shared__ unsigned short Al[Hdim * LDS_STRIDE];  // Al[r][k] = bf16(A[row0+r][k])
  const int tid  = threadIdx.x;
  const int row0 = blockIdx.x * 128;

  // cooperative load + f32->bf16 convert into LDS
  for (int i = tid; i < Hdim * Hdim / 4; i += 256) {
    int r  = i >> 5;
    int c4 = (i & 31) << 2;
    float4 wv = *(const float4*)(W + r * Hdim + c4);
    *(unsigned int*)&Wl[r * LDS_STRIDE + c4]     = f2bf(wv.x) | (f2bf(wv.y) << 16);
    *(unsigned int*)&Wl[r * LDS_STRIDE + c4 + 2] = f2bf(wv.z) | (f2bf(wv.w) << 16);

    int row = row0 + r;
    float4 av;
    if (row < N) av = *(const float4*)(A + (size_t)row * Hdim + c4);
    else { av.x = av.y = av.z = av.w = 0.0f; }
    *(unsigned int*)&Al[r * LDS_STRIDE + c4]     = f2bf(av.x) | (f2bf(av.y) << 16);
    *(unsigned int*)&Al[r * LDS_STRIDE + c4 + 2] = f2bf(av.z) | (f2bf(av.w) << 16);
  }
  __syncthreads();

  const int wave = tid >> 5;        // 0..7 : 16-row strip
  const int lane = tid & 31;
  const int lr   = lane & 15;
  const int hi   = lane >> 4;
  const int m0   = wave * 16;

  v8f acc[8];
  for (int n = 0; n < 8; ++n)
    for (int i = 0; i < 8; ++i) acc[n][i] = 0.0f;

  #pragma unroll
  for (int kk = 0; kk < 4; ++kk) {        // K chunks of 32
    const int kb = kk * 32;
    // A fragment: 16x32 bf16 (lanes 0-15: K 0..15, 16-31: K 16..31)
    BFrag af;
    #pragma unroll
    for (int i = 0; i < 8; ++i) {
      int kloc = 2 * i + hi * 16;
      af.u[i] = *(const unsigned int*)&Al[(m0 + lr) * LDS_STRIDE + kb + kloc];
    }
    // B fragments: 32x16 bf16 (lane = column, lanes 0-15: K 0..15, 16-31: K 16..31)
    // double-buffered: load tile n+1 while WMMA consumes tile n
    BFrag bb[2];
    #pragma unroll
    for (int i = 0; i < 8; ++i) {
      int kloc = 2 * i + hi * 16;
      bb[0].u[i] = *(const unsigned int*)&Wl[lr * LDS_STRIDE + kb + kloc];
    }
    #pragma unroll
    for (int n = 0; n < 8; ++n) {
      if (n < 7) {
        #pragma unroll
        for (int i = 0; i < 8; ++i) {
          int kloc = 2 * i + hi * 16;
          bb[(n + 1) & 1].u[i] =
              *(const unsigned int*)&Wl[((n + 1) * 16 + lr) * LDS_STRIDE + kb + kloc];
        }
      }
      acc[n] = __builtin_amdgcn_wmma_f32_16x16x32_bf16(
          false, af.v, false, bb[n & 1].v, (short)0, acc[n], false, false);
    }
  }

  // epilogue: bias (+resid), f32 store
  #pragma unroll
  for (int n = 0; n < 8; ++n) {
    int col = n * 16 + lr;
    float bv = bias ? bias[col] : 0.0f;
    #pragma unroll
    for (int v = 0; v < 8; ++v) {
      int row = row0 + m0 + hi * 8 + v;
      if (row < N) {
        float val = acc[n][v] + bv;
        if (resid) val += resid[(size_t)row * Hdim + col];
        C[(size_t)row * Hdim + col] = val;
      }
    }
  }
}

// ---------------------------------------------------------------------------
// e_new = Ce + Ah[ei1] + Bh[ei0]; agg[ei0] += sigmoid(e_new) * Vh[ei1]
// ---------------------------------------------------------------------------
__global__ void edge_combine_kernel(const float* __restrict__ Ah, const float* __restrict__ Bh,
                                    const float* __restrict__ Vh,
                                    const int* __restrict__ ei0, const int* __restrict__ ei1,
                                    float* __restrict__ ebuf, float* __restrict__ agg, int E)
{
  long long idx = (long long)blockIdx.x * blockDim.x + threadIdx.x;
  if (idx >= (long long)E * Hdim) return;
  int e = (int)(idx >> 7);
  int c = (int)(idx & 127);
  int s0 = ei0[e], s1 = ei1[e];
  float en = ebuf[idx] + Ah[(size_t)s1 * Hdim + c] + Bh[(size_t)s0 * Hdim + c];
  float g  = 1.0f / (1.0f + __expf(-en));
  atomicAdd(&agg[(size_t)s0 * Hdim + c], g * Vh[(size_t)s1 * Hdim + c]);
  ebuf[idx] = en;
}

// ---------------------------------------------------------------------------
// h_out = h_in + relu(LN(Uh + agg))         (one row per wave32)
// ---------------------------------------------------------------------------
__global__ __launch_bounds__(256)
void node_update_kernel(const float* __restrict__ Uh, const float* __restrict__ agg,
                        const float* __restrict__ h_in,
                        const float* __restrict__ g, const float* __restrict__ b,
                        float* __restrict__ h_out, int V)
{
  int row  = blockIdx.x * 8 + (threadIdx.x >> 5);
  int lane = threadIdx.x & 31;
  if (row >= V) return;
  size_t base = (size_t)row * Hdim;
  float x[4]; float s = 0.f, sq = 0.f;
  #pragma unroll
  for (int j = 0; j < 4; ++j) {
    int c = lane + 32 * j;
    float v = Uh[base + c] + agg[base + c];
    x[j] = v; s += v; sq += v * v;
  }
  #pragma unroll
  for (int off = 16; off >= 1; off >>= 1) {
    s  += __shfl_xor(s,  off, 32);
    sq += __shfl_xor(sq, off, 32);
  }
  float m  = s  * (1.0f / 128.0f);
  float vr = sq * (1.0f / 128.0f) - m * m;
  float r  = rsqrtf(vr + 1e-5f);
  #pragma unroll
  for (int j = 0; j < 4; ++j) {
    int c = lane + 32 * j;
    float y = (x[j] - m) * r * g[c] + b[c];
    y = y > 0.f ? y : 0.f;
    h_out[base + c] = h_in[base + c] + y;
  }
}

// ---------------------------------------------------------------------------
// in-place on ebuf:  y = relu(LN_e(e_new)) + tbias ;  ebuf = silu(LN_plo(y))
// ---------------------------------------------------------------------------
__global__ __launch_bounds__(256)
void edge_update_kernel(float* __restrict__ ebuf,
                        const float* __restrict__ lng, const float* __restrict__ lnb,
                        const float* __restrict__ plog, const float* __restrict__ plob,
                        const float* __restrict__ tbias, int E)
{
  int row  = blockIdx.x * 8 + (threadIdx.x >> 5);
  int lane = threadIdx.x & 31;
  if (row >= E) return;
  size_t base = (size_t)row * Hdim;
  float x[4]; float s = 0.f, sq = 0.f;
  #pragma unroll
  for (int j = 0; j < 4; ++j) {
    int c = lane + 32 * j;
    float v = ebuf[base + c];
    x[j] = v; s += v; sq += v * v;
  }
  #pragma unroll
  for (int off = 16; off >= 1; off >>= 1) { s += __shfl_xor(s, off, 32); sq += __shfl_xor(sq, off, 32); }
  float m = s * (1.0f / 128.0f);
  float r = rsqrtf(sq * (1.0f / 128.0f) - m * m + 1e-5f);
  s = 0.f; sq = 0.f;
  #pragma unroll
  for (int j = 0; j < 4; ++j) {
    int c = lane + 32 * j;
    float y = (x[j] - m) * r * lng[c] + lnb[c];
    y = y > 0.f ? y : 0.f;
    y += tbias[c];
    x[j] = y; s += y; sq += y * y;
  }
  #pragma unroll
  for (int off = 16; off >= 1; off >>= 1) { s += __shfl_xor(s, off, 32); sq += __shfl_xor(sq, off, 32); }
  m = s * (1.0f / 128.0f);
  r = rsqrtf(sq * (1.0f / 128.0f) - m * m + 1e-5f);
  #pragma unroll
  for (int j = 0; j < 4; ++j) {
    int c = lane + 32 * j;
    float y = (x[j] - m) * r * plog[c] + plob[c];
    float sg = 1.0f / (1.0f + __expf(-y));
    ebuf[base + c] = y * sg;     // silu
  }
}

// ---------------------------------------------------------------------------
// Embedding feature builders
// ---------------------------------------------------------------------------
__global__ void posfeat_kernel(const float* __restrict__ x, float* __restrict__ feat, int V)
{
  int idx = blockIdx.x * blockDim.x + threadIdx.x;
  if (idx >= V * Hdim) return;
  int v = idx >> 7, f = idx & 127;
  int coord = f >> 6;                 // 0 -> x, 1 -> y
  int ff = f & 63;
  int j  = ff >> 1;
  float dim_t = __powf(10000.0f, (float)j * (1.0f / 32.0f));
  float p = x[v * 2 + coord] * 6.283185307179586f / dim_t;
  feat[idx] = (ff & 1) ? __cosf(p) : __sinf(p);
}

__global__ void edgefeat_kernel(const float* __restrict__ graph, float* __restrict__ feat, int E)
{
  long long idx = (long long)blockIdx.x * blockDim.x + threadIdx.x;
  if (idx >= (long long)E * Hdim) return;
  int e = (int)(idx >> 7), f = (int)(idx & 127);
  int j = f >> 1;
  float dim_t = __powf(10000.0f, (float)j * (1.0f / 64.0f));
  float p = graph[e] / dim_t;
  feat[idx] = (f & 1) ? __cosf(p) : __sinf(p);
}

// time embedding + MLP + per-layer bias:  tbias[l] = relu(te2) @ tl_w[l].T + tl_b[l]
__global__ __launch_bounds__(128)
void time_kernel(const float* __restrict__ t,
                 const float* __restrict__ w1, const float* __restrict__ b1,
                 const float* __restrict__ w2, const float* __restrict__ b2,
                 const float* __restrict__ tlw, const float* __restrict__ tlb,
                 float* __restrict__ tbias)
{
  __shared__ float te[Hdim], t1[Hdim], r2[Hdim];
  int j = threadIdx.x;
  float tt = t[0];
  {
    int half = j & 63;
    float fr = __expf(-logf(10000.0f) * (float)half * (1.0f / 64.0f));
    float a  = tt * fr;
    te[j] = (j < 64) ? __cosf(a) : __sinf(a);   // concat([cos, sin])
  }
  __syncthreads();
  float s = b1[j];
  for (int k = 0; k < Hdim; ++k) s += w1[j * Hdim + k] * te[k];
  t1[j] = s > 0.f ? s : 0.f;
  __syncthreads();
  s = b2[j];
  for (int k = 0; k < Hdim; ++k) s += w2[j * Hdim + k] * t1[k];
  r2[j] = s > 0.f ? s : 0.f;                    // relu(te2)
  __syncthreads();
  for (int l = 0; l < 3; ++l) {
    float o = tlb[l * Hdim + j];
    for (int k = 0; k < Hdim; ++k) o += tlw[l * Hdim * Hdim + j * Hdim + k] * r2[k];
    tbias[l * Hdim + j] = o;
  }
}

__global__ void zero_kernel(float* __restrict__ p, int n)
{
  int i = blockIdx.x * blockDim.x + threadIdx.x;
  if (i < n) p[i] = 0.0f;
}

// out[e] = dot(e_row, out_w) + out_b   (one row per wave32)
__global__ __launch_bounds__(256)
void out_kernel(const float* __restrict__ e, const float* __restrict__ w,
                const float* __restrict__ b, float* __restrict__ out, int E)
{
  int row  = blockIdx.x * 8 + (threadIdx.x >> 5);
  int lane = threadIdx.x & 31;
  if (row >= E) return;
  size_t base = (size_t)row * Hdim;
  float s = 0.f;
  #pragma unroll
  for (int j = 0; j < 4; ++j) { int c = lane + 32 * j; s += e[base + c] * w[c]; }
  #pragma unroll
  for (int off = 16; off >= 1; off >>= 1) s += __shfl_xor(s, off, 32);
  if (lane == 0) out[row] = s + b[0];
}

// ---------------------------------------------------------------------------
extern "C" void kernel_launch(void* const* d_in, const int* in_sizes, int n_in,
                              void* d_out, int out_size, void* d_ws, size_t ws_size,
                              hipStream_t stream)
{
  const float* x     = (const float*)d_in[0];
  const float* graph = (const float*)d_in[1];
  const float* ts    = (const float*)d_in[2];
  const int*   eidx  = (const int*)d_in[3];

  enum { NEW_=0, NEB, EEW, EEB, TW1, TB1, TW2, TB2,
         UW, UB, VW, VB, AW, AB, BW, BB, CW, CB,
         LNHG, LNHB, LNEG, LNEB, TLW, TLB, PLOG, PLOB, PLOW, PLOBIAS, OUTW, OUTB };
  const float* P[30];
  for (int i = 0; i < 30; ++i) P[i] = (const float*)d_in[4 + i];

  const int V = in_sizes[0] / 2;
  const int E = in_sizes[1];
  const int HH = Hdim * Hdim;
  const size_t VH = (size_t)V * Hdim;
  const size_t EH = (size_t)E * Hdim;

  float* ws     = (float*)d_ws;
  float* f_h0   = ws;
  float* f_h1   = f_h0 + VH;
  float* f_Uh   = f_h1 + VH;
  float* f_Vh   = f_Uh + VH;
  float* f_Ah   = f_Vh + VH;
  float* f_Bh   = f_Ah + VH;
  float* f_agg  = f_Bh + VH;
  float* f_feat = f_agg + VH;
  float* f_tb   = f_feat + VH;      // 3*128
  float* f_e0   = f_tb + 3 * Hdim;
  float* f_e1   = f_e0 + EH;
  float* f_ebuf = f_e1 + EH;

  const int* ei0 = eidx;        // edge_index[0]
  const int* ei1 = eidx + E;    // edge_index[1]

  const dim3 blk256(256), blk128(128);
  const int gN   = (V + 127) / 128;            // node GEMM tiles
  const int gE   = (E + 127) / 128;            // edge GEMM tiles
  const int gVe  = (int)((VH + 255) / 256);    // node elementwise
  const int gEe  = (int)((EH + 255) / 256);    // edge elementwise
  const int gVr  = (V + 7) / 8;                // node row/wave
  const int gEr  = (E + 7) / 8;                // edge row/wave

  // time conditioning (tiny)
  time_kernel<<<1, blk128, 0, stream>>>(ts, P[TW1], P[TB1], P[TW2], P[TB2],
                                        P[TLW], P[TLB], f_tb);
  // node embedding: h = pos_embed @ node_embed_w.T + b
  posfeat_kernel<<<gVe, blk256, 0, stream>>>(x, f_feat, V);
  gemm_wmma<<<gN, blk256, 0, stream>>>(f_feat, P[NEW_], P[NEB], nullptr, f_h0, V);
  // edge embedding: e = scalar_embed @ edge_embed_w.T + b
  edgefeat_kernel<<<gEe, blk256, 0, stream>>>(graph, f_ebuf, E);
  gemm_wmma<<<gE, blk256, 0, stream>>>(f_ebuf, P[EEW], P[EEB], nullptr, f_e0, E);

  float* h_cur = f_h0; float* h_nxt = f_h1;
  float* e_cur = f_e0; float* e_nxt = f_e1;

  for (int l = 0; l < 3; ++l) {
    const float* Uw = P[UW] + l * HH; const float* Ub = P[UB] + l * Hdim;
    const float* Vw = P[VW] + l * HH; const float* Vb = P[VB] + l * Hdim;
    const float* Aw = P[AW] + l * HH; const float* Ab = P[AB] + l * Hdim;
    const float* Bw = P[BW] + l * HH; const float* Bb = P[BB] + l * Hdim;
    const float* Cw = P[CW] + l * HH; const float* Cb = P[CB] + l * Hdim;

    gemm_wmma<<<gN, blk256, 0, stream>>>(h_cur, Uw, Ub, nullptr, f_Uh, V);
    gemm_wmma<<<gN, blk256, 0, stream>>>(h_cur, Vw, Vb, nullptr, f_Vh, V);
    gemm_wmma<<<gN, blk256, 0, stream>>>(h_cur, Aw, Ab, nullptr, f_Ah, V);
    gemm_wmma<<<gN, blk256, 0, stream>>>(h_cur, Bw, Bb, nullptr, f_Bh, V);
    gemm_wmma<<<gE, blk256, 0, stream>>>(e_cur, Cw, Cb, nullptr, f_ebuf, E);

    zero_kernel<<<gVe, blk256, 0, stream>>>(f_agg, (int)VH);
    edge_combine_kernel<<<gEe, blk256, 0, stream>>>(f_Ah, f_Bh, f_Vh, ei0, ei1,
                                                    f_ebuf, f_agg, E);
    node_update_kernel<<<gVr, blk256, 0, stream>>>(f_Uh, f_agg, h_cur,
                                                   P[LNHG] + l * Hdim, P[LNHB] + l * Hdim,
                                                   h_nxt, V);
    edge_update_kernel<<<gEr, blk256, 0, stream>>>(f_ebuf,
                                                   P[LNEG] + l * Hdim, P[LNEB] + l * Hdim,
                                                   P[PLOG] + l * Hdim, P[PLOB] + l * Hdim,
                                                   f_tb + l * Hdim, E);
    // e_nxt = e_in + silu_branch @ plo_w.T + plo_bias   (residual fused)
    gemm_wmma<<<gE, blk256, 0, stream>>>(f_ebuf, P[PLOW] + l * HH, P[PLOBIAS] + l * Hdim,
                                         e_cur, e_nxt, E);

    float* t;
    t = h_cur; h_cur = h_nxt; h_nxt = t;
    t = e_cur; e_cur = e_nxt; e_nxt = t;
  }

  out_kernel<<<gEr, blk256, 0, stream>>>(e_cur, P[OUTW], P[OUTB], (float*)d_out, E);
}
